// TrainableGATLayer_6047313953575
// MI455X (gfx1250) — matrix operand order repaired and checked by reading, not compiled, not run on hardware
//
#include <hip/hip_runtime.h>
#include <hip/hip_bf16.h>

typedef __attribute__((ext_vector_type(16))) __bf16 v16bf;
typedef __attribute__((ext_vector_type(8)))  float  v8f;

#define IN_DIM  128
#define OUT_DIM 128
#define NEG_SLOPE 0.2f

__device__ __forceinline__ void atomicAddF(float* p, float v) {
    __hip_atomic_fetch_add(p, v, __ATOMIC_RELAXED, __HIP_MEMORY_SCOPE_AGENT);
}

// monotone float -> uint mapping so unsigned atomicMax == float max
__device__ __forceinline__ unsigned f2ord(float f) {
    unsigned u = __float_as_uint(f);
    return (u & 0x80000000u) ? ~u : (u | 0x80000000u);
}
__device__ __forceinline__ float ord2f(unsigned u) {
    unsigned b = (u & 0x80000000u) ? (u & 0x7fffffffu) : ~u;
    return __uint_as_float(b);
}

// ---------------------------------------------------------------------------
// Kernel 0: out = bias (broadcast), mord = 0 (== -inf in ordered encoding),
// denom = 0. Must run every call: harness poisons buffers.
// ---------------------------------------------------------------------------
__global__ void gat_init(float* __restrict__ out, const float* __restrict__ bias,
                         unsigned* __restrict__ mord, float* __restrict__ denom, int N) {
    int i = blockIdx.x * blockDim.x + threadIdx.x;
    int total = N * OUT_DIM;
    if (i < total) out[i] = bias[i & (OUT_DIM - 1)];
    if (i < N) { mord[i] = 0u; denom[i] = 0.0f; }
}

// ---------------------------------------------------------------------------
// Kernel 1: h = x @ W via bf16-split WMMA (fp32-accurate), fused a_s/a_d.
// Block = 128 threads = 4 waves, each wave owns one 16-row tile and all 128
// output columns (8 column tiles of 16). W is split hi/lo bf16, transposed
// into LDS so B fragments are contiguous 32B ds loads matching the
// contiguous k order (k = 16*half + s) of the A fragments.
// ---------------------------------------------------------------------------
__global__ __launch_bounds__(128) void gat_gemm(
    const float* __restrict__ x, const float* __restrict__ W,
    const float* __restrict__ att_src, const float* __restrict__ att_dst,
    float* __restrict__ h, float* __restrict__ a_s, float* __restrict__ a_d, int N)
{
    __shared__ __align__(32) __bf16 WT_hi[IN_DIM * OUT_DIM];  // [n][k], 32 KB
    __shared__ __align__(32) __bf16 WT_lo[IN_DIM * OUT_DIM];  // [n][k], 32 KB

    const int t = threadIdx.x;            // 0..127
    // Fill LDS: thread t owns row k=t of W (128 consecutive floats).
    {
        const float* wrow = W + (size_t)t * OUT_DIM;
        #pragma unroll 4
        for (int n = 0; n < OUT_DIM; ++n) {
            float wf = wrow[n];
            __bf16 wh = (__bf16)wf;
            float rs = wf - (float)wh;
            WT_hi[n * IN_DIM + t] = wh;
            WT_lo[n * IN_DIM + t] = (__bf16)rs;
        }
    }
    __syncthreads();

    const int wave = t >> 5;
    const int lane = t & 31;
    const int half = lane >> 4;           // 0: lanes 0-15, 1: lanes 16-31
    const int nl   = lane & 15;
    const int rowBase = (blockIdx.x * 4 + wave) * 16;
    if (rowBase >= N) return;             // wave-uniform: EXEC stays all-ones

    // A fragments for this lane: row M = nl; K chunks in contiguous order (k = 16*half + s):
    // elems 0..7  -> K = q*32 + half*16 + i
    // elems 8..15 -> K = q*32 + half*16 + 8 + i
    const float* xrow = x + (size_t)(rowBase + nl) * IN_DIM;
    v16bf Ah[4], Al[4];
    #pragma unroll
    for (int q = 0; q < 4; ++q) {
        #pragma unroll
        for (int i = 0; i < 8; ++i) {
            float f0 = xrow[q * 32 + half * 16 + i];
            __bf16 h0 = (__bf16)f0;
            Ah[q][i] = h0;
            Al[q][i] = (__bf16)(f0 - (float)h0);
            float f1 = xrow[q * 32 + half * 16 + 8 + i];
            __bf16 h1 = (__bf16)f1;
            Ah[q][i + 8] = h1;
            Al[q][i + 8] = (__bf16)(f1 - (float)h1);
        }
    }

    float psum[8], pdum[8];
    #pragma unroll
    for (int v = 0; v < 8; ++v) { psum[v] = 0.0f; pdum[v] = 0.0f; }

    for (int j = 0; j < 8; ++j) {
        v8f acc = {};
        const float sA = att_src[j * 16 + nl];
        const float sD = att_dst[j * 16 + nl];
        const int bbase = (j * 16 + nl) * IN_DIM + half * 16;
        #pragma unroll
        for (int q = 0; q < 4; ++q) {
            const v16bf Bh = *(const v16bf*)(WT_hi + bbase + q * 32);
            const v16bf Bl = *(const v16bf*)(WT_lo + bbase + q * 32);
            acc = __builtin_amdgcn_wmma_f32_16x16x32_bf16(false, Ah[q], false, Bh,
                                                          (short)0, acc, false, false);
            acc = __builtin_amdgcn_wmma_f32_16x16x32_bf16(false, Al[q], false, Bh,
                                                          (short)0, acc, false, false);
            acc = __builtin_amdgcn_wmma_f32_16x16x32_bf16(false, Ah[q], false, Bl,
                                                          (short)0, acc, false, false);
        }
        // C/D layout: elem v -> row M = v + half*8, col N = nl
        #pragma unroll
        for (int v = 0; v < 8; ++v) {
            float hv = acc[v];
            h[(size_t)(rowBase + v + half * 8) * OUT_DIM + j * 16 + nl] = hv;
            psum[v] += hv * sA;
            pdum[v] += hv * sD;
        }
    }

    // Reduce attention partials across the 16 lanes holding one row's columns.
    #pragma unroll
    for (int v = 0; v < 8; ++v) {
        float s = psum[v], d = pdum[v];
        #pragma unroll
        for (int mk = 8; mk >= 1; mk >>= 1) {
            s += __shfl_xor(s, mk, 32);
            d += __shfl_xor(d, mk, 32);
        }
        if (nl == 0) {
            a_s[rowBase + v + half * 8] = s;
            a_d[rowBase + v + half * 8] = d;
        }
    }
}

// ---------------------------------------------------------------------------
// Kernel 2: segment max over edge scores via ordered-uint atomicMax.
// Edges e < E come from edge_index; e >= E are self loops (src=dst=e-E).
// ---------------------------------------------------------------------------
__global__ void gat_edge_max(const int* __restrict__ ei,
                             const float* __restrict__ a_s, const float* __restrict__ a_d,
                             unsigned* __restrict__ mord, int E, int Etot) {
    int e = blockIdx.x * blockDim.x + threadIdx.x;
    if (e >= Etot) return;
    int s, d;
    if (e < E) { s = ei[e]; d = ei[E + e]; } else { s = d = e - E; }
    float v = a_s[s] + a_d[d];
    v = (v >= 0.0f) ? v : NEG_SLOPE * v;
    atomicMax(mord + d, f2ord(v));
}

// ---------------------------------------------------------------------------
// Kernel 3: p = exp(e - m[dst]); denom[dst] += p (hw fp32 atomic).
// ---------------------------------------------------------------------------
__global__ void gat_edge_exp(const int* __restrict__ ei,
                             const float* __restrict__ a_s, const float* __restrict__ a_d,
                             const unsigned* __restrict__ mord,
                             float* __restrict__ p, float* __restrict__ denom,
                             int E, int Etot) {
    int e = blockIdx.x * blockDim.x + threadIdx.x;
    if (e >= Etot) return;
    int s, d;
    if (e < E) { s = ei[e]; d = ei[E + e]; } else { s = d = e - E; }
    float v = a_s[s] + a_d[d];
    v = (v >= 0.0f) ? v : NEG_SLOPE * v;
    float pe = __expf(v - ord2f(mord[d]));
    p[e] = pe;
    atomicAddF(denom + d, pe);
}

// ---------------------------------------------------------------------------
// Kernel 4: out[dst] += alpha * h[src]. One wave per edge, float4 per lane.
// This is the bandwidth-dominant pass (~850 MB total traffic).
// ---------------------------------------------------------------------------
__global__ __launch_bounds__(256) void gat_scatter(
    const int* __restrict__ ei, const float* __restrict__ h,
    const float* __restrict__ p, const float* __restrict__ denom,
    float* __restrict__ out, int E, int Etot)
{
    int wid = blockIdx.x * 8 + (threadIdx.x >> 5);
    if (wid >= Etot) return;
    int lane = threadIdx.x & 31;
    int s, d;
    if (wid < E) { s = ei[wid]; d = ei[E + wid]; } else { s = d = wid - E; }
    float alpha = p[wid] / denom[d];             // uniform loads, coalesce
    float4 hv = *(const float4*)(h + (size_t)s * OUT_DIM + lane * 4);
    float* op = out + (size_t)d * OUT_DIM + lane * 4;
    atomicAddF(op + 0, hv.x * alpha);
    atomicAddF(op + 1, hv.y * alpha);
    atomicAddF(op + 2, hv.z * alpha);
    atomicAddF(op + 3, hv.w * alpha);
}

// ---------------------------------------------------------------------------
extern "C" void kernel_launch(void* const* d_in, const int* in_sizes, int n_in,
                              void* d_out, int out_size, void* d_ws, size_t ws_size,
                              hipStream_t stream) {
    const float* x       = (const float*)d_in[0];
    const int*   ei      = (const int*)  d_in[1];   // edge_index flat [2*E]
    const float* W       = (const float*)d_in[2];
    const float* att_src = (const float*)d_in[3];
    const float* att_dst = (const float*)d_in[4];
    const float* bias    = (const float*)d_in[5];

    const int N    = in_sizes[0] / IN_DIM;
    const int E    = in_sizes[1] / 2;
    const int Etot = E + N;                          // + self loops

    float* out = (float*)d_out;

    // workspace layout (all 4B types, d_ws assumed aligned)
    float*    h     = (float*)d_ws;                  // N*128
    float*    a_s   = h + (size_t)N * OUT_DIM;       // N
    float*    a_d   = a_s + N;                       // N
    unsigned* mord  = (unsigned*)(a_d + N);          // N
    float*    denom = (float*)(mord + N);            // N
    float*    p     = denom + N;                     // Etot

    {
        int total = N * OUT_DIM;
        gat_init<<<(total + 255) / 256, 256, 0, stream>>>(out, bias, mord, denom, N);
    }
    {
        int rowTiles = N / 16;                       // N = 50000 = 3125 * 16
        gat_gemm<<<(rowTiles + 3) / 4, 128, 0, stream>>>(x, W, att_src, att_dst,
                                                         h, a_s, a_d, N);
    }
    gat_edge_max<<<(Etot + 255) / 256, 256, 0, stream>>>(ei, a_s, a_d, mord, E, Etot);
    gat_edge_exp<<<(Etot + 255) / 256, 256, 0, stream>>>(ei, a_s, a_d, mord, p, denom, E, Etot);
    gat_scatter<<<(Etot + 7) / 8, 256, 0, stream>>>(ei, h, p, denom, out, E, Etot);
}
